// CLAM_Graph_SB_42984032698884
// MI455X (gfx1250) — matrix operand order repaired and checked by reading, not compiled, not run on hardware
//
#include <hip/hip_runtime.h>
#include <hip/hip_bf16.h>

#define NN 20000
#define NE 320000
#define LL 512
#define DD 256

typedef __attribute__((ext_vector_type(16))) _Float16 v16h;
typedef __attribute__((ext_vector_type(8)))  _Float16 v8h;
typedef __attribute__((ext_vector_type(8)))  float    v8f;

// ---------- ordered-int float max helpers (monotone map for atomicMax) ----------
static __device__ __forceinline__ unsigned f2ord(float x) {
  unsigned b = __float_as_uint(x);
  return (b & 0x80000000u) ? ~b : (b | 0x80000000u);
}
static __device__ __forceinline__ float ord2f(unsigned u) {
  unsigned b = (u & 0x80000000u) ? (u & 0x7fffffffu) : ~u;
  return __uint_as_float(b);
}
#define ORD_NEG_INF 0x007FFFFFu

static __device__ __forceinline__ float waveSum(float v) {
  #pragma unroll
  for (int m = 16; m > 0; m >>= 1) v += __shfl_xor(v, m, 32);
  return v;
}

// ---------------- prep kernels ----------------
__global__ void k_conv_x(const float* __restrict__ x, _Float16* __restrict__ xh, int n) {
  for (int i = blockIdx.x * blockDim.x + threadIdx.x; i < n; i += gridDim.x * blockDim.x)
    xh[i] = (_Float16)x[i];
}

// Wt[n*512 + k] = Wcat[k][n] (f16), n in [0,1024) across {Wq,Wk,Wv,Ws}; bcat[n]
__global__ void k_prep_w(const float* __restrict__ Wq, const float* __restrict__ Wk,
                         const float* __restrict__ Wv, const float* __restrict__ Ws,
                         const float* __restrict__ bq, const float* __restrict__ bk,
                         const float* __restrict__ bv, const float* __restrict__ bs,
                         _Float16* __restrict__ wt, float* __restrict__ bcat) {
  int total = 1024 * LL;
  for (int i = blockIdx.x * blockDim.x + threadIdx.x; i < total; i += gridDim.x * blockDim.x) {
    int n = i >> 9;          // 0..1023
    int k = i & 511;         // 0..511
    int mat = n >> 8;
    int nn = n & 255;
    const float* W = (mat == 0) ? Wq : (mat == 1) ? Wk : (mat == 2) ? Wv : Ws;
    wt[i] = (_Float16)W[(size_t)k * DD + nn];
    if (i < 1024) {
      const float* b = (i < 256) ? bq : (i < 512) ? bk : (i < 768) ? bv : bs;
      bcat[i] = b[i & 255];
    }
  }
}

// Wa1t[n*256 + k] = Wa1[k*256 + n] (f16)
__global__ void k_prep_wa1(const float* __restrict__ Wa1, _Float16* __restrict__ wa1t) {
  int total = DD * DD;
  for (int i = blockIdx.x * blockDim.x + threadIdx.x; i < total; i += gridDim.x * blockDim.x) {
    int n = i >> 8, k = i & 255;
    wa1t[i] = (_Float16)Wa1[(size_t)k * DD + n];
  }
}

__global__ void k_init(float* __restrict__ agg, float* __restrict__ den,
                       unsigned* __restrict__ m_u, float* __restrict__ Mnum,
                       float* __restrict__ den2, unsigned* __restrict__ rmax) {
  int total = NN * DD;
  for (int i = blockIdx.x * blockDim.x + threadIdx.x; i < total; i += gridDim.x * blockDim.x) {
    agg[i] = 0.0f;
    if (i < NN) { den[i] = 0.0f; m_u[i] = ORD_NEG_INF; }
    if (i < 2 * DD) Mnum[i] = 0.0f;
    if (i < 2) { den2[i] = 0.0f; rmax[i] = ORD_NEG_INF; }
  }
}

// ---------------- WMMA GEMM ----------------
// Fragment loaders for 16-bit A / B; both use contiguous k order (k = 16*(lane/16) + s).
static __device__ __forceinline__ v16h loadFragA(const _Float16* __restrict__ p) {
  // per lane: 16 consecutive halves
  v8h lo = *(const v8h*)p;
  v8h hi = *(const v8h*)(p + 8);
  v16h a;
  #pragma unroll
  for (int i = 0; i < 8; ++i) { a[i] = lo[i]; a[8 + i] = hi[i]; }
  return a;
}
static __device__ __forceinline__ v16h loadFragB(const _Float16* __restrict__ p) {
  // per lane: 16 consecutive halves
  v8h lo = *(const v8h*)p;
  v8h hi = *(const v8h*)(p + 8);
  v16h b;
  #pragma unroll
  for (int i = 0; i < 8; ++i) { b[i] = lo[i]; b[8 + i] = hi[i]; }
  return b;
}

// A: M x K (f16, row-major). Bt: N x K (f16, row-major = B column-major).
// Out: split into 256-wide matrices (mat = globalN >> 8), each M x 256 f32.
// Each wave owns one 16-row M-tile and 8 consecutive 16-col N-tiles.
// Rotating software pipeline: B fragment t for stage kk+32 is loaded right after
// WMMA t of stage kk consumes the slot (~8 WMMAs of latency cover per load);
// A is double-buffered. ~160 live VGPRs; launch_bounds widens the budget so
// nothing spills to scratch.
__global__ void __launch_bounds__(256, 1)
k_gemm_wmma(const _Float16* __restrict__ A, const _Float16* __restrict__ Bt,
            const float* __restrict__ bias, float* __restrict__ Out,
            int M, int K, int nGroups) {
  int wave = blockIdx.x * (blockDim.x >> 5) + (threadIdx.x >> 5);
  int mTiles = M >> 4;
  if (wave >= mTiles * nGroups) return;
  int mTile = wave / nGroups;
  int nGroup = wave - mTile * nGroups;
  int lane = threadIdx.x & 31;
  int r  = lane & 15;
  int ph = lane >> 4;

  // A fragment source: lane<16 -> row r, K kk..kk+15;
  //                    lane>=16 -> row r, K kk+16..kk+31
  const _Float16* arow = A + (size_t)(mTile * 16 + r) * K + ph * 16;
  const _Float16* brow[8];
  #pragma unroll
  for (int t = 0; t < 8; ++t) {
    int nTile = nGroup * 8 + t;
    // B fragment: lane<16 -> col (nTile*16+r), K kk..kk+15; lane>=16 -> K kk+16..kk+31
    brow[t] = Bt + (size_t)(nTile * 16 + r) * K + ph * 16;
  }

  v8f acc[8] = {};

  // stage 0 preload
  v16h a_cur = loadFragA(arow);
  v16h b_cur[8];
  #pragma unroll
  for (int t = 0; t < 8; ++t) b_cur[t] = loadFragB(brow[t]);

  for (int kk = 0; kk < K; kk += 32) {
    v16h a_nxt = loadFragA(arow + kk + 32);          // next-stage A (tail read stays in ws)
    __builtin_prefetch(arow + kk + 64, 0, 0);        // global_prefetch_b8, one stage ahead

    #pragma unroll
    for (int t = 0; t < 8; ++t) {
      acc[t] = __builtin_amdgcn_wmma_f32_16x16x32_f16(
          false, a_cur, false, b_cur[t], (short)0, acc[t], false, false);
      b_cur[t] = loadFragB(brow[t] + kk + 32);       // refill slot for next stage
    }
    a_cur = a_nxt;
  }

  // C/D layout: VGPR rr, lanes 0-15: (M=rr, N=lane); lanes 16-31: (M=rr+8, N=lane-16)
  #pragma unroll
  for (int t = 0; t < 8; ++t) {
    int colIn = (nGroup * 8 + t) * 16 + r;
    int mat = colIn >> 8;
    int col = colIn & 255;
    float bv = bias[colIn];
    float* outBase = Out + (size_t)mat * M * DD;
    #pragma unroll
    for (int rr = 0; rr < 8; ++rr) {
      int row = mTile * 16 + rr + ph * 8;
      outBase[(size_t)row * DD + col] = acc[t][rr] + bv;
    }
  }
}

// ---------------- edge kernels ----------------
// score[e] = dot(q[dst], k[src]) / 16 ; segment max via ordered-int atomicMax
__global__ void k_edge_score(const float* __restrict__ q, const float* __restrict__ k,
                             const int* __restrict__ src, const int* __restrict__ dst,
                             float* __restrict__ score, unsigned* __restrict__ m_u) {
  int e = blockIdx.x * (blockDim.x >> 5) + (threadIdx.x >> 5);
  if (e >= NE) return;
  int lane = threadIdx.x & 31;
  int s = src[e], d = dst[e];
  const float4* qp = (const float4*)(q + (size_t)d * DD) + lane * 2;
  const float4* kp = (const float4*)(k + (size_t)s * DD) + lane * 2;
  float4 q0 = qp[0], q1 = qp[1];
  float4 k0 = kp[0], k1 = kp[1];
  float acc = q0.x * k0.x + q0.y * k0.y + q0.z * k0.z + q0.w * k0.w
            + q1.x * k1.x + q1.y * k1.y + q1.z * k1.z + q1.w * k1.w;
  acc = waveSum(acc) * (1.0f / 16.0f);
  if (lane == 0) {
    score[e] = acc;
    atomicMax(&m_u[d], f2ord(acc));
  }
}

__global__ void k_edge_exp(const float* __restrict__ score, const int* __restrict__ dst,
                           const unsigned* __restrict__ m_u, float* __restrict__ den,
                           float* __restrict__ eex) {
  int e = blockIdx.x * blockDim.x + threadIdx.x;
  if (e >= NE) return;
  int d = dst[e];
  float ee = __expf(score[e] - ord2f(m_u[d]));
  eex[e] = ee;
  atomicAdd(&den[d], ee);
}

// agg[dst] += alpha * v[src]
__global__ void k_edge_agg(const float* __restrict__ eex, const float* __restrict__ den,
                           const float* __restrict__ vmat, const int* __restrict__ src,
                           const int* __restrict__ dst, float* __restrict__ agg) {
  int e = blockIdx.x * (blockDim.x >> 5) + (threadIdx.x >> 5);
  if (e >= NE) return;
  int lane = threadIdx.x & 31;
  int s = src[e], d = dst[e];
  float a = eex[e] / den[d];
  const float* vrow = vmat + (size_t)s * DD + lane * 8;
  float* arow = agg + (size_t)d * DD + lane * 8;
  float4 v0 = ((const float4*)vrow)[0];
  float4 v1 = ((const float4*)vrow)[1];
  atomicAdd(arow + 0, a * v0.x); atomicAdd(arow + 1, a * v0.y);
  atomicAdd(arow + 2, a * v0.z); atomicAdd(arow + 3, a * v0.w);
  atomicAdd(arow + 4, a * v1.x); atomicAdd(arow + 5, a * v1.y);
  atomicAdd(arow + 6, a * v1.z); atomicAdd(arow + 7, a * v1.w);
}

// h = gelu_exact(agg + sproj); write f32 h and f16 h
__global__ void k_gelu(const float* __restrict__ agg, const float* __restrict__ smat,
                       float* __restrict__ h, _Float16* __restrict__ hh) {
  int total = NN * DD;
  for (int i = blockIdx.x * blockDim.x + threadIdx.x; i < total; i += gridDim.x * blockDim.x) {
    float hv = agg[i] + smat[i];
    float g = 0.5f * hv * (1.0f + erff(hv * 0.70710678118654752f));
    h[i] = g;
    hh[i] = (_Float16)g;
  }
}

// scores2[n][c] = tanh(u[n]) . Wa2[:,c] + ba2[c]; global max via atomicMax
__global__ void k_node_score(const float* __restrict__ u, const float* __restrict__ Wa2,
                             const float* __restrict__ ba2, float* __restrict__ s2,
                             unsigned* __restrict__ rmax) {
  int n = blockIdx.x * (blockDim.x >> 5) + (threadIdx.x >> 5);
  if (n >= NN) return;
  int lane = threadIdx.x & 31;
  const float* urow = u + (size_t)n * DD + lane * 8;
  float p0 = 0.0f, p1 = 0.0f;
  #pragma unroll
  for (int j = 0; j < 8; ++j) {
    int d = lane * 8 + j;
    float t = tanhf(urow[j]);
    p0 += t * Wa2[d * 2 + 0];
    p1 += t * Wa2[d * 2 + 1];
  }
  p0 = waveSum(p0);
  p1 = waveSum(p1);
  if (lane == 0) {
    float s0 = p0 + ba2[0], s1 = p1 + ba2[1];
    s2[n * 2 + 0] = s0;
    s2[n * 2 + 1] = s1;
    atomicMax(&rmax[0], f2ord(s0));
    atomicMax(&rmax[1], f2ord(s1));
  }
}

// den2[c] += e_c ; Mnum[c][d] += e_c * h[n][d]  (LDS-staged partials per block)
__global__ void k_node_M(const float* __restrict__ s2, const unsigned* __restrict__ rmax,
                         const float* __restrict__ h, float* __restrict__ den2,
                         float* __restrict__ Mnum) {
  __shared__ float part[2 * DD];
  __shared__ float pden[2];
  for (int i = threadIdx.x; i < 2 * DD; i += blockDim.x) part[i] = 0.0f;
  if (threadIdx.x < 2) pden[threadIdx.x] = 0.0f;
  __syncthreads();

  int n = blockIdx.x * (blockDim.x >> 5) + (threadIdx.x >> 5);
  int lane = threadIdx.x & 31;
  if (n < NN) {
    float m0 = ord2f(rmax[0]), m1 = ord2f(rmax[1]);
    float e0 = __expf(s2[n * 2 + 0] - m0);
    float e1 = __expf(s2[n * 2 + 1] - m1);
    if (lane == 0) { atomicAdd(&pden[0], e0); atomicAdd(&pden[1], e1); }
    const float* hrow = h + (size_t)n * DD + lane * 8;
    #pragma unroll
    for (int j = 0; j < 8; ++j) {
      int d = lane * 8 + j;
      float hv = hrow[j];
      atomicAdd(&part[d], e0 * hv);
      atomicAdd(&part[DD + d], e1 * hv);
    }
  }
  __syncthreads();
  for (int i = threadIdx.x; i < 2 * DD; i += blockDim.x) atomicAdd(&Mnum[i], part[i]);
  if (threadIdx.x < 2) atomicAdd(&den2[threadIdx.x], pden[threadIdx.x]);
}

// logits, Y_prob (single wave)
__global__ void k_final(const float* __restrict__ Mnum, const float* __restrict__ den2,
                        const float* __restrict__ Wc, const float* __restrict__ bc,
                        float* __restrict__ out) {
  int lane = threadIdx.x & 31;
  float p0 = 0.0f, p1 = 0.0f;
  #pragma unroll
  for (int j = 0; j < 8; ++j) {
    int d = lane * 8 + j;
    p0 += Mnum[d] * Wc[d];
    p1 += Mnum[DD + d] * Wc[DD + d];
  }
  p0 = waveSum(p0);
  p1 = waveSum(p1);
  if (lane == 0) {
    float l0 = p0 / den2[0] + bc[0];
    float l1 = p1 / den2[1] + bc[1];
    out[0] = l0; out[1] = l1;
    float mx = fmaxf(l0, l1);
    float z0 = __expf(l0 - mx), z1 = __expf(l1 - mx);
    float zs = z0 + z1;
    out[2] = z0 / zs; out[3] = z1 / zs;
  }
}

// A[n] = attn[n][label]
__global__ void k_outA(const float* __restrict__ s2, const unsigned* __restrict__ rmax,
                       const float* __restrict__ den2, const int* __restrict__ label,
                       float* __restrict__ outA) {
  int lab = label[0];
  float m = ord2f(rmax[lab]);
  float dn = den2[lab];
  for (int n = blockIdx.x * blockDim.x + threadIdx.x; n < NN; n += gridDim.x * blockDim.x)
    outA[n] = __expf(s2[n * 2 + lab] - m) / dn;
}

// ---------------- workspace layout (bytes) ----------------
#define OFF_XH    ((size_t)0)                         // 20000*512 f16
#define OFF_WT    (OFF_XH   + (size_t)NN*LL*2)        // 1024*512 f16
#define OFF_BCAT  (OFF_WT   + (size_t)1024*LL*2)      // 1024 f32
#define OFF_O     (OFF_BCAT + (size_t)1024*4)         // 4 * 20000*256 f32 (q,k,v,s)
#define OFF_SCORE (OFF_O    + (size_t)4*NN*DD*4)      // NE f32
#define OFF_EEX   (OFF_SCORE+ (size_t)NE*4)           // NE f32
#define OFF_MU    (OFF_EEX  + (size_t)NE*4)           // NN u32
#define OFF_DEN   (OFF_MU   + (size_t)NN*4)           // NN f32
#define OFF_AGG   (OFF_DEN  + (size_t)NN*4)           // NN*DD f32
#define OFF_WA1T  (OFF_AGG  + (size_t)NN*DD*4)        // 256*256 f16
#define OFF_S2    (OFF_WA1T + (size_t)DD*DD*2)        // NN*2 f32
#define OFF_RED   (OFF_S2   + (size_t)NN*2*4)         // rmax[2] u32, den2[2] f32, Mnum[512] f32

extern "C" void kernel_launch(void* const* d_in, const int* in_sizes, int n_in,
                              void* d_out, int out_size, void* d_ws, size_t ws_size,
                              hipStream_t stream) {
  const float* x    = (const float*)d_in[0];
  const int*   eidx = (const int*)d_in[1];
  const int*   lab  = (const int*)d_in[2];
  const float* Wq = (const float*)d_in[3],  *bq = (const float*)d_in[4];
  const float* Wk = (const float*)d_in[5],  *bk = (const float*)d_in[6];
  const float* Wv = (const float*)d_in[7],  *bv = (const float*)d_in[8];
  const float* Ws = (const float*)d_in[9],  *bs = (const float*)d_in[10];
  const float* Wa1 = (const float*)d_in[11], *ba1 = (const float*)d_in[12];
  const float* Wa2 = (const float*)d_in[13], *ba2 = (const float*)d_in[14];
  const float* Wc  = (const float*)d_in[15], *bc  = (const float*)d_in[16];
  const int* src = eidx;
  const int* dst = eidx + NE;

  char* ws = (char*)d_ws;
  _Float16* xh   = (_Float16*)(ws + OFF_XH);
  _Float16* wt   = (_Float16*)(ws + OFF_WT);
  float*    bcat = (float*)(ws + OFF_BCAT);
  float*    O    = (float*)(ws + OFF_O);        // q | k | v | s (contiguous)
  float*    qm   = O;
  float*    km   = O + (size_t)NN * DD;
  float*    vm   = O + (size_t)2 * NN * DD;
  float*    sm   = O + (size_t)3 * NN * DD;
  float*    score= (float*)(ws + OFF_SCORE);
  float*    eex  = (float*)(ws + OFF_EEX);
  unsigned* m_u  = (unsigned*)(ws + OFF_MU);
  float*    den  = (float*)(ws + OFF_DEN);
  float*    agg  = (float*)(ws + OFF_AGG);
  _Float16* wa1t = (_Float16*)(ws + OFF_WA1T);
  float*    s2   = (float*)(ws + OFF_S2);
  unsigned* rmax = (unsigned*)(ws + OFF_RED);
  float*    den2 = (float*)(ws + OFF_RED + 8);
  float*    Mnum = (float*)(ws + OFF_RED + 16);
  // aliases of dead projection buffers:
  float*    hmat = qm;                 // h (f32) overwrites q
  float*    umat = km;                 // u = tanh-input overwrites k
  _Float16* hh   = (_Float16*)vm;      // h (f16) overwrites v

  float* out = (float*)d_out;

  k_conv_x   <<<2048, 256, 0, stream>>>(x, xh, NN * LL);
  k_prep_w   <<<2048, 256, 0, stream>>>(Wq, Wk, Wv, Ws, bq, bk, bv, bs, wt, bcat);
  k_prep_wa1 <<<256, 256, 0, stream>>>(Wa1, wa1t);
  k_init     <<<2048, 256, 0, stream>>>(agg, den, m_u, Mnum, den2, rmax);

  // q,k,v,s = x @ [Wq|Wk|Wv|Ws] + b : 1250 mTiles * 8 nGroups = 10000 waves
  k_gemm_wmma<<<1250, 256, 0, stream>>>(xh, wt, bcat, O, NN, LL, 8);

  k_edge_score<<<NE / 8, 256, 0, stream>>>(qm, km, src, dst, score, m_u);
  k_edge_exp  <<<(NE + 255) / 256, 256, 0, stream>>>(score, dst, m_u, den, eex);
  k_edge_agg  <<<NE / 8, 256, 0, stream>>>(eex, den, vm, src, dst, agg);

  k_gelu<<<2048, 256, 0, stream>>>(agg, sm, hmat, hh);

  // u = h @ Wa1 + ba1 : 1250 mTiles * 2 nGroups = 2500 waves
  k_gemm_wmma<<<313, 256, 0, stream>>>(hh, wa1t, ba1, umat, NN, DD, 2);

  k_node_score<<<NN / 8, 256, 0, stream>>>(umat, Wa2, ba2, s2, rmax);
  k_node_M    <<<NN / 8, 256, 0, stream>>>(s2, rmax, hmat, den2, Mnum);
  k_final     <<<1, 32, 0, stream>>>(Mnum, den2, Wc, bc, out);
  k_outA      <<<128, 256, 0, stream>>>(s2, rmax, den2, lab, out + 4);
}
